// MutiGAT_36636071035352
// MI455X (gfx1250) — matrix operand order repaired and checked by reading, not compiled, not run on hardware
//
#include <hip/hip_runtime.h>
#include <hip/hip_bf16.h>
#include <math.h>

#define TPB 256
#define NF  128   // feature dim (K of the big GEMM)

typedef __attribute__((ext_vector_type(16))) __bf16 v16bf;
typedef __attribute__((ext_vector_type(8)))  __bf16 v8bf;
typedef __attribute__((ext_vector_type(8)))  float  v8f;

// ---------- helpers ----------
__device__ __forceinline__ unsigned f32_enc(float f) {
  unsigned u = __float_as_uint(f);
  return (u & 0x80000000u) ? ~u : (u | 0x80000000u);
}
__device__ __forceinline__ float f32_dec(unsigned e) {
  unsigned u = (e & 0x80000000u) ? (e & 0x7fffffffu) : ~e;
  return __uint_as_float(u);
}
__device__ __forceinline__ void edge_sd(const int* ei, int E, int N, int e, int& s, int& d) {
  if (e < E) { s = ei[e]; d = ei[E + e]; }     // (2,E) row-major
  else       { s = e - E; d = e - E; }          // self loop
}
__device__ __forceinline__ float sigm(float x) { return 1.f / (1.f + expf(-x)); }

// ---------- tiny utility kernels ----------
__global__ void zero_f32_kernel(float* p, int n) {
  int i = blockIdx.x * blockDim.x + threadIdx.x;
  if (i < n) p[i] = 0.f;
}
__global__ void fill_u32_kernel(unsigned* p, unsigned v, int n) {
  int i = blockIdx.x * blockDim.x + threadIdx.x;
  if (i < n) p[i] = v;
}
__global__ void mask_count_kernel(const int* mask, float* cnt, int N) {
  int i = blockIdx.x * blockDim.x + threadIdx.x;
  if (i < N && mask[i] > 0) atomicAdd(cnt, 1.f);
}

// ---------- GAT stage ----------
__global__ void gemm_xw_kernel(const float* __restrict__ x, const float* __restrict__ W,
                               float* __restrict__ h, int N, int F, int K) {
  int idx = blockIdx.x * blockDim.x + threadIdx.x;
  if (idx >= N * K) return;
  int n = idx / K, k = idx % K;
  const float* xr = x + (size_t)n * F;
  float acc = 0.f;
  for (int f = 0; f < F; ++f) acc = fmaf(xr[f], W[(size_t)f * K + k], acc);
  h[idx] = acc;
}
__global__ void att_node_kernel(const float* __restrict__ h, const float* __restrict__ as,
                                const float* __restrict__ ad, float* al, float* ar, int N, int K) {
  int n = blockIdx.x * blockDim.x + threadIdx.x;
  if (n >= N) return;
  const float* hr = h + (size_t)n * K;
  float l = 0.f, r = 0.f;
  for (int k = 0; k < K; ++k) { l = fmaf(hr[k], as[k], l); r = fmaf(hr[k], ad[k], r); }
  al[n] = l; ar[n] = r;
}
__global__ void edge_max_kernel(const int* ei, int E, int N, const float* al, const float* ar,
                                unsigned* menc) {
  int e = blockIdx.x * blockDim.x + threadIdx.x;
  if (e >= E + N) return;
  int s, d; edge_sd(ei, E, N, e, s, d);
  float v = al[s] + ar[d];
  v = (v >= 0.f) ? v : 0.2f * v;
  atomicMax(&menc[d], f32_enc(v));
}
__global__ void edge_expsum_kernel(const int* ei, int E, int N, const float* al, const float* ar,
                                   const unsigned* menc, float* ee, float* ssum) {
  int e = blockIdx.x * blockDim.x + threadIdx.x;
  if (e >= E + N) return;
  int s, d; edge_sd(ei, E, N, e, s, d);
  float v = al[s] + ar[d];
  v = (v >= 0.f) ? v : 0.2f * v;
  float w = expf(v - f32_dec(menc[d]));
  ee[e] = w;
  atomicAdd(&ssum[d], w);
}
__global__ void edge_agg_kernel(const int* ei, int E, int N, const float* ee, const float* ssum,
                                const float* h, float* agg, int K) {
  long long idx = (long long)blockIdx.x * blockDim.x + threadIdx.x;
  long long total = (long long)(E + N) * K;
  if (idx >= total) return;
  int e = (int)(idx / K), k = (int)(idx % K);
  int s, d; edge_sd(ei, E, N, e, s, d);
  float alpha = ee[e] / (ssum[d] + 1e-16f);
  atomicAdd(&agg[(size_t)d * K + k], alpha * h[(size_t)s * K + k]);
}
__global__ void node_post1_kernel(float* hid, const float* b, const float* gamma,
                                  const float* beta, int N, int K) {
  int idx = blockIdx.x * blockDim.x + threadIdx.x;
  if (idx >= N * K) return;
  int k = idx % K;
  float inv = rsqrtf(1.0f + 1e-5f);
  float v = (hid[idx] + b[k]) * gamma[k] * inv + beta[k];
  hid[idx] = (v >= 0.f) ? v : 0.01f * v;
}
__global__ void layer2_node_kernel(const float* __restrict__ hid, const float* __restrict__ W2,
                                   const float* as2, const float* ad2,
                                   float* h2, float* al2, float* ar2, int N) {
  int n = blockIdx.x * blockDim.x + threadIdx.x;
  if (n >= N) return;
  const float* hr = hid + (size_t)n * 64;
  float c0 = 0.f, c1 = 0.f;
  for (int k = 0; k < 64; ++k) {
    float hv = hr[k];
    c0 = fmaf(hv, W2[k * 2 + 0], c0);
    c1 = fmaf(hv, W2[k * 2 + 1], c1);
  }
  h2[n * 2] = c0; h2[n * 2 + 1] = c1;
  al2[n] = c0 * as2[0] + c1 * as2[1];
  ar2[n] = c0 * ad2[0] + c1 * ad2[1];
}
__global__ void loss_out_kernel(const float* o, const float* b2, const int* y, const int* mask,
                                float* outacc, float* lossacc, int N) {
  int n = blockIdx.x * blockDim.x + threadIdx.x;
  if (n >= N) return;
  float o0 = o[n * 2] + b2[0], o1 = o[n * 2 + 1] + b2[1];
  float mx = fmaxf(o0, o1);
  float lse = mx + logf(expf(o0 - mx) + expf(o1 - mx));
  outacc[n] += expf(o1 - lse);
  if (mask[n] > 0) {
    float lpy = ((y[n] == 1) ? o1 : o0) - lse;
    atomicAdd(lossacc, -lpy);
  }
}

// ---------- dense stage prep ----------
__global__ void nodep_kernel(const float* outacc, float* nodep, float* out0, int N, int G) {
  int n = blockIdx.x * blockDim.x + threadIdx.x;
  if (n >= N) return;
  float v = outacc[n] / (float)G;
  nodep[n] = v;
  out0[n] = v;   // node_p output
}
__global__ void prep_ex_kernel(const float* x, const float* emb_w, const float* logit_p,
                               float* ex, __bf16* exhi, __bf16* exlo, float* pw, int N, int F) {
  int idx = blockIdx.x * blockDim.x + threadIdx.x;
  if (idx >= N * F) return;
  int f = idx % F;
  const float EPSf = 2.220446049250313e-16f;
  float p = sigm(logit_p[f]);
  float approx = logf(p + EPSf) - logf(1.f - p + EPSf);   // log(0.5)-log(0.5) cancels
  float keep = 1.f - sigm(approx * 10.f);
  float e = x[idx] * keep * sigm(emb_w[f]);
  ex[idx] = e;
  __bf16 hi = (__bf16)e;
  exhi[idx] = hi;
  exlo[idx] = (__bf16)(e - (float)hi);
  if (idx < F) pw[idx] = 1.f - sigm(logit_p[idx]);        // pw output
}
// store RECIPROCAL so the WMMA epilogue is a multiply, not an IEEE divide
__global__ void rowdenominv_kernel(const float* ex, float* denominv, int N, int F) {
  int n = blockIdx.x * blockDim.x + threadIdx.x;
  if (n >= N) return;
  const float* r = ex + (size_t)n * F;
  float s = 0.f;
  for (int f = 0; f < F; ++f) s += r[f];
  denominv[n] = 1.f / (s + 1e-6f);
}

// ---------- fused WMMA GEMM: graph = ex @ ex.T * denominv[col], + fused row reductions ----------
// Block = 8 waves = 128 rows; the 8 waves SHARE one 16-col B tile staged in LDS
// (double-buffered, filled with global_load_async_to_lds_b128 on ASYNCcnt).
__device__ __forceinline__ void async_copy_tile(const __bf16* gh, const __bf16* gl,
                                                __bf16* sh, __bf16* sl, int Jbase, int t) {
  const int c  = t >> 4;          // 0..15 column within tile
  const int ch = t & 15;          // 0..15 16-byte chunk along K
  const size_t off = (size_t)(Jbase + c) * NF + ch * 8;
  const int    loff = (c * NF + ch * 8) * 2;    // bytes
  unsigned dh = (unsigned)(size_t)sh + (unsigned)loff;
  unsigned dl = (unsigned)(size_t)sl + (unsigned)loff;
  unsigned long long ah = (unsigned long long)(size_t)(gh + off);
  unsigned long long al = (unsigned long long)(size_t)(gl + off);
  asm volatile("global_load_async_to_lds_b128 %0, %1, off" :: "v"(dh), "v"(ah) : "memory");
  asm volatile("global_load_async_to_lds_b128 %0, %1, off" :: "v"(dl), "v"(al) : "memory");
}

__global__ __launch_bounds__(256)
void graph_wmma_kernel(const __bf16* __restrict__ exhi, const __bf16* __restrict__ exlo,
                       const float* __restrict__ denominv, const float* __restrict__ nodep,
                       float* __restrict__ graph, float* __restrict__ rowsum,
                       float* __restrict__ adj, int N) {
  const int lane    = threadIdx.x & 31;
  const int wave    = threadIdx.x >> 5;            // 0..7 -> 16-row strip
  const int half    = lane >> 4;
  const int l15     = lane & 15;
  const int rowbase = blockIdx.x * 128 + wave * 16;
  const int njt     = N / 16;
  // N % 16 == 0, so a strip is either fully in-range or fully padded: wave-uniform.
  const bool rowsValid = (rowbase + 16) <= N;

  __shared__ __attribute__((aligned(32))) __bf16 s_bh[2][16 * NF];
  __shared__ __attribute__((aligned(32))) __bf16 s_bl[2][16 * NF];

  // A fragments for this wave's 16-row strip (rows may be in the zero-padded tail).
  v16bf ahi[4], alo[4];
  {
    const int row = rowbase + l15;
    const __bf16* ph = exhi + (size_t)row * NF;
    const __bf16* pl = exlo + (size_t)row * NF;
#pragma unroll
    for (int kk = 0; kk < 4; ++kk) {
      const int k0 = kk * 32 + half * 16;       // elems 0..7
      const int k1 = kk * 32 + half * 16 + 8;   // elems 8..15
      v8bf h0 = *(const v8bf*)(ph + k0);
      v8bf h1 = *(const v8bf*)(ph + k1);
      ahi[kk] = __builtin_shufflevector(h0, h1, 0,1,2,3,4,5,6,7,8,9,10,11,12,13,14,15);
      v8bf l0 = *(const v8bf*)(pl + k0);
      v8bf l1 = *(const v8bf*)(pl + k1);
      alo[kk] = __builtin_shufflevector(l0, l1, 0,1,2,3,4,5,6,7,8,9,10,11,12,13,14,15);
    }
  }

  float rs[8], ap[8];
#pragma unroll
  for (int r = 0; r < 8; ++r) { rs[r] = 0.f; ap[r] = 0.f; }

  // prologue: stage tile 0
  async_copy_tile(exhi, exlo, s_bh[0], s_bl[0], 0, threadIdx.x);
  asm volatile("s_wait_asynccnt 0" ::: "memory");
  __syncthreads();

  for (int jt = 0; jt < njt; ++jt) {
    const int cur = jt & 1;
    if (jt + 1 < njt)
      async_copy_tile(exhi, exlo, s_bh[cur ^ 1], s_bl[cur ^ 1], (jt + 1) * 16, threadIdx.x);

    const int Jbase = jt * 16;
    const int col = Jbase + l15;
    const __bf16* bh = &s_bh[cur][l15 * NF];
    const __bf16* bl = &s_bl[cur][l15 * NF];
    // two independent accumulator chains for XDL pipelining
    v8f acc0 = {0.f, 0.f, 0.f, 0.f, 0.f, 0.f, 0.f, 0.f};
    v8f acc1 = {0.f, 0.f, 0.f, 0.f, 0.f, 0.f, 0.f, 0.f};
#pragma unroll
    for (int kk = 0; kk < 4; ++kk) {
      const int koff = kk * 32 + half * 16;     // 16 consecutive K per lane-half
      v16bf bhi = *(const v16bf*)(bh + koff);
      v16bf blo = *(const v16bf*)(bl + koff);
      acc0 = __builtin_amdgcn_wmma_f32_16x16x32_bf16(false, ahi[kk], false, bhi, (short)0, acc0, false, false);
      acc1 = __builtin_amdgcn_wmma_f32_16x16x32_bf16(false, ahi[kk], false, blo, (short)0, acc1, false, false);
      acc1 = __builtin_amdgcn_wmma_f32_16x16x32_bf16(false, alo[kk], false, bhi, (short)0, acc1, false, false);
    }
    if (rowsValid) {            // wave-uniform: one s_cbranch, no per-store exec juggling
      const float dinv  = denominv[col];
      const float npcol = nodep[col];
#pragma unroll
      for (int r = 0; r < 8; ++r) {
        const int row = rowbase + half * 8 + r; // C/D layout: lanes 16-31 hold M=r+8
        const float v = (acc0[r] + acc1[r]) * dinv;
        // 400MB write-once stream: NT hint keeps the 5MB B matrix resident in L2
        __builtin_nontemporal_store(v, &graph[(size_t)row * N + col]);
        rs[r] += v;
        ap[r] += v * npcol;
      }
    }
    asm volatile("s_wait_asynccnt 0" ::: "memory");
    __syncthreads();
  }

  // each wave owns its 16 rows exclusively: shuffle-reduce across the 16 column lanes
  if (rowsValid) {
#pragma unroll
    for (int r = 0; r < 8; ++r) {
      float a = rs[r], b = ap[r];
      for (int off = 8; off >= 1; off >>= 1) {
        a += __shfl_xor(a, off, 32);
        b += __shfl_xor(b, off, 32);
      }
      const int row = rowbase + half * 8 + r;
      if (l15 == 0) { rowsum[row] = a; adj[row] = b; }
    }
  }
}

__global__ void finalize_kernel(const float* lossacc, const float* maskcnt,
                                const float* rowsum, const float* adj,
                                float* out, int N, int G) {
  int n = blockIdx.x * blockDim.x + threadIdx.x;
  if (n >= N) return;
  // norm[:,1] = sigmoid(adj - adj_n) = sigmoid(2*adj - rowsum)
  float z = 2.f * adj[n] - rowsum[n];
  out[N + 1 + n] = 1.f / (1.f + expf(-z));
  if (n == 0) out[N] = lossacc[0] / (fmaxf(maskcnt[0], 1.f) * (float)G);
}

// ---------- launcher ----------
static inline dim3 gridFor(long long n) { return dim3((unsigned)((n + TPB - 1) / TPB)); }

extern "C" void kernel_launch(void* const* d_in, const int* in_sizes, int n_in,
                              void* d_out, int out_size, void* d_ws, size_t ws_size,
                              hipStream_t stream) {
  const int N = in_sizes[1];           // 10000
  const int F = in_sizes[16];          // 128
  const int H = 64, C = 2, G = 4;
  const int E = in_sizes[3] / 2;       // 320000
  const int Etot = E + N;
  const int Npad = ((N + 127) / 128) * 128;   // rows padded for 128-row macro-tiles

  const float* x      = (const float*)d_in[0];
  const int*   y      = (const int*)  d_in[1];
  const int*   tmask  = (const int*)  d_in[2];
  const int*   e_ppi  = (const int*)  d_in[3];
  const int*   e_hom  = (const int*)  d_in[4];
  const float* W1     = (const float*)d_in[5];
  const float* a_src1 = (const float*)d_in[6];
  const float* a_dst1 = (const float*)d_in[7];
  const float* b1     = (const float*)d_in[8];
  const float* gamma  = (const float*)d_in[9];
  const float* beta   = (const float*)d_in[10];
  const float* W2     = (const float*)d_in[11];
  const float* a_src2 = (const float*)d_in[12];
  const float* a_dst2 = (const float*)d_in[13];
  const float* b2     = (const float*)d_in[14];
  const float* emb_w  = (const float*)d_in[15];
  const float* logitp = (const float*)d_in[16];

  float* out_f = (float*)d_out;
  const size_t OUT_GRAPH = (size_t)(2 * N + 1);
  const size_t OUT_PW    = OUT_GRAPH + (size_t)N * N;

  // bump allocator over workspace
  char* wp = (char*)d_ws;
  auto alloc = [&](size_t bytes) -> char* {
    char* p = wp;
    wp += (bytes + 255) & ~(size_t)255;
    return p;
  };
  float*    h1      = (float*)   alloc((size_t)N * H * 4);
  float*    hid     = (float*)   alloc((size_t)N * H * 4);
  float*    al      = (float*)   alloc((size_t)N * 4);
  float*    ar      = (float*)   alloc((size_t)N * 4);
  unsigned* menc    = (unsigned*)alloc((size_t)N * 4);
  float*    ssum    = (float*)   alloc((size_t)N * 4);
  float*    ee      = (float*)   alloc((size_t)Etot * 4);
  float*    h2      = (float*)   alloc((size_t)N * C * 4);
  float*    o2      = (float*)   alloc((size_t)N * C * 4);
  float*    al2     = (float*)   alloc((size_t)N * 4);
  float*    ar2     = (float*)   alloc((size_t)N * 4);
  float*    outacc  = (float*)   alloc((size_t)N * 4);
  float*    lossacc = (float*)   alloc(256);
  float*    maskcnt = (float*)   alloc(256);
  float*    ex      = (float*)   alloc((size_t)N * F * 4);
  __bf16*   exhi    = (__bf16*)  alloc((size_t)Npad * F * 2);
  __bf16*   exlo    = (__bf16*)  alloc((size_t)Npad * F * 2);
  float*    denominv= (float*)   alloc((size_t)N * 4);
  float*    nodep   = (float*)   alloc((size_t)N * 4);
  float*    rowsum  = (float*)   alloc((size_t)N * 4);
  float*    adjv    = (float*)   alloc((size_t)N * 4);

  const unsigned ENC_NEG_INF = 0x007FFFFFu;   // f32_enc(-inf)

  zero_f32_kernel<<<gridFor(N), TPB, 0, stream>>>(outacc, N);
  zero_f32_kernel<<<1, TPB, 0, stream>>>(lossacc, 2);
  zero_f32_kernel<<<1, TPB, 0, stream>>>(maskcnt, 2);
  mask_count_kernel<<<gridFor(N), TPB, 0, stream>>>(tmask, maskcnt, N);

  for (int g = 0; g < G; ++g) {
    const int* ei = (g % 2 == 0) ? e_ppi : e_hom;
    const float* W1g = W1 + (size_t)g * F * H;
    const float* as1 = a_src1 + (size_t)g * H;
    const float* ad1 = a_dst1 + (size_t)g * H;
    const float* b1g = b1 + (size_t)g * H;
    const float* gmg = gamma + (size_t)g * H;
    const float* btg = beta + (size_t)g * H;
    const float* W2g = W2 + (size_t)g * H * C;
    const float* as2 = a_src2 + (size_t)g * C;
    const float* ad2 = a_dst2 + (size_t)g * C;
    const float* b2g = b2 + (size_t)g * C;

    // ---- layer 1 ----
    fill_u32_kernel<<<gridFor(N), TPB, 0, stream>>>(menc, ENC_NEG_INF, N);
    zero_f32_kernel<<<gridFor(N), TPB, 0, stream>>>(ssum, N);
    zero_f32_kernel<<<gridFor((long long)N * H), TPB, 0, stream>>>(hid, N * H);
    gemm_xw_kernel<<<gridFor((long long)N * H), TPB, 0, stream>>>(x, W1g, h1, N, F, H);
    att_node_kernel<<<gridFor(N), TPB, 0, stream>>>(h1, as1, ad1, al, ar, N, H);
    edge_max_kernel<<<gridFor(Etot), TPB, 0, stream>>>(ei, E, N, al, ar, menc);
    edge_expsum_kernel<<<gridFor(Etot), TPB, 0, stream>>>(ei, E, N, al, ar, menc, ee, ssum);
    edge_agg_kernel<<<gridFor((long long)Etot * H), TPB, 0, stream>>>(ei, E, N, ee, ssum, h1, hid, H);
    node_post1_kernel<<<gridFor((long long)N * H), TPB, 0, stream>>>(hid, b1g, gmg, btg, N, H);

    // ---- layer 2 ----
    fill_u32_kernel<<<gridFor(N), TPB, 0, stream>>>(menc, ENC_NEG_INF, N);
    zero_f32_kernel<<<gridFor(N), TPB, 0, stream>>>(ssum, N);
    zero_f32_kernel<<<gridFor((long long)N * C), TPB, 0, stream>>>(o2, N * C);
    layer2_node_kernel<<<gridFor(N), TPB, 0, stream>>>(hid, W2g, as2, ad2, h2, al2, ar2, N);
    edge_max_kernel<<<gridFor(Etot), TPB, 0, stream>>>(ei, E, N, al2, ar2, menc);
    edge_expsum_kernel<<<gridFor(Etot), TPB, 0, stream>>>(ei, E, N, al2, ar2, menc, ee, ssum);
    edge_agg_kernel<<<gridFor((long long)Etot * C), TPB, 0, stream>>>(ei, E, N, ee, ssum, h2, o2, C);
    loss_out_kernel<<<gridFor(N), TPB, 0, stream>>>(o2, b2g, y, tmask, outacc, lossacc, N);
  }

  // ---- dense stage ----
  nodep_kernel<<<gridFor(N), TPB, 0, stream>>>(outacc, nodep, out_f, N, G);
  prep_ex_kernel<<<gridFor((long long)N * F), TPB, 0, stream>>>(x, emb_w, logitp, ex, exhi, exlo,
                                                                out_f + OUT_PW, N, F);
  // zero the padded tail rows of exhi/exlo (bf16 zeros) so padded-row WMMAs are benign
  {
    int padWords = (Npad - N) * F / 2;   // bf16 pairs as u32
    if (padWords > 0) {
      fill_u32_kernel<<<gridFor(padWords), TPB, 0, stream>>>(
          (unsigned*)(exhi + (size_t)N * F), 0u, padWords);
      fill_u32_kernel<<<gridFor(padWords), TPB, 0, stream>>>(
          (unsigned*)(exlo + (size_t)N * F), 0u, padWords);
    }
  }
  rowdenominv_kernel<<<gridFor(N), TPB, 0, stream>>>(ex, denominv, N, F);
  graph_wmma_kernel<<<dim3(Npad / 128), 256, 0, stream>>>(exhi, exlo, denominv, nodep,
                                                          out_f + OUT_GRAPH, rowsum, adjv, N);
  finalize_kernel<<<gridFor(N), TPB, 0, stream>>>(lossacc, maskcnt, rowsum, adjv, out_f, N, G);
  (void)n_in; (void)out_size; (void)ws_size;
}
